// NAIS_regionEmbedding_78400333021191
// MI455X (gfx1250) — matrix operand order repaired and checked by reading, not compiled, not run on hardware
//
#include <hip/hip_runtime.h>
#include <cmath>

typedef _Float16 f16;
typedef __attribute__((ext_vector_type(16))) _Float16 v16h;
typedef __attribute__((ext_vector_type(8)))  float    v8f;

#define NB    1024
#define DEMB  64
#define TWO_D 128
#define HID   256

// Dynamic LDS layout (bytes):
//   [0,      65536) : W1^T as f16, [n=256][k=128]
//   [65536,  66048) : tgt_i row, f32 [128]
//   [66048,  67072) : b1, f32 [256]
//   [67072,  68096) : W2, f32 [256]
//   [68096,  68224) : reduction scratch, f32 [16][2]
#define LDS_BYTES 68224

extern "C" __global__ __launch_bounds__(256)
void nais_region_kernel(const int* __restrict__ history,
                        const int* __restrict__ target,
                        const int* __restrict__ history_region,
                        const int* __restrict__ target_region,
                        const float* __restrict__ emb_history,
                        const float* __restrict__ emb_target,
                        const float* __restrict__ emb_region,
                        const float* __restrict__ W1,
                        const float* __restrict__ b1,
                        const float* __restrict__ W2,
                        float* __restrict__ out)
{
    extern __shared__ char smem[];
    f16*   w1t  = (f16*)smem;                  // [256][128], w1t[n*128+k] = W1[k][n]
    float* tgts = (float*)(smem + 65536);      // [128]
    float* b1s  = (float*)(smem + 66048);      // [256]
    float* w2s  = (float*)(smem + 67072);      // [256]
    float* red  = (float*)(smem + 68096);      // [32]

    const int i = blockIdx.x;       // target row this workgroup owns
    const int t = threadIdx.x;

    // ---- stage W1^T (f16), tgt_i row, b1, W2 into LDS ----
    for (int idx = t; idx < HID * TWO_D; idx += 256) {
        const int n = idx >> 7;          // hidden unit
        const int k = idx & 127;         // input feature
        w1t[idx] = (f16)W1[k * HID + n];
    }
    const int ti = target[i];
    const int tr = target_region[i];
    if (t < TWO_D) {
        tgts[t] = (t < DEMB) ? emb_target[(size_t)ti * DEMB + t]
                             : emb_region[(size_t)tr * DEMB + (t - DEMB)];
    }
    if (t < HID) { b1s[t] = b1[t]; w2s[t] = W2[t]; }
    __syncthreads();

    const int wid  = t >> 5;
    const int lane = t & 31;
    const int ln   = lane & 15;   // column / row-within-halfwave index
    const int hs   = lane >> 4;   // half-wave select

    // B fragment with ones in column 0 only: row-sum extractor (dot product).
    v16h bones;
    {
        const f16 v = (ln == 0) ? (f16)1.0f : (f16)0.0f;
        #pragma unroll
        for (int q = 0; q < 16; ++q) bones[q] = v;
    }

    float s0 = 0.0f, s1 = 0.0f;   // only meaningful on lanes 0 and 16

    for (int pass = 0; pass < 8; ++pass) {
        // This lane supplies A-matrix row M = ln (both half-waves load same row,
        // different K ranges per the 16-bit A 16x32 layout).
        const int jrow = pass * 128 + wid * 16 + ln;
        const int ih = history[jrow];
        const int ir = history_region[jrow];
        const float* hrow = emb_history + (size_t)ih * DEMB;
        const float* rrow = emb_region  + (size_t)ir * DEMB;

        // ---- build 4 A fragments: A[j,k] = hist[j,k] * tgt_i[k], f16 ----
        v16h afr[4];
        #pragma unroll
        for (int ks = 0; ks < 4; ++ks) {
            const int k0 = ks * 32 + hs * 8;   // first 8-chunk  (K = k0..k0+7)
            const int k1 = k0 + 16;            // second 8-chunk (K = k1..k1+7)
            const float* p0 = (k0 < DEMB) ? (hrow + k0) : (rrow + (k0 - DEMB));
            const float* p1 = (k1 < DEMB) ? (hrow + k1) : (rrow + (k1 - DEMB));
            const float4* g0 = (const float4*)p0;
            const float4* g1 = (const float4*)p1;
            const float4* t0 = (const float4*)(tgts + k0);
            const float4* t1 = (const float4*)(tgts + k1);
            float4 ha = g0[0], hb = g0[1], hc = g1[0], hd = g1[1];
            float4 ta = t0[0], tb = t0[1], tc = t1[0], td = t1[1];
            afr[ks][0]  = (f16)(ha.x * ta.x); afr[ks][1]  = (f16)(ha.y * ta.y);
            afr[ks][2]  = (f16)(ha.z * ta.z); afr[ks][3]  = (f16)(ha.w * ta.w);
            afr[ks][4]  = (f16)(hb.x * tb.x); afr[ks][5]  = (f16)(hb.y * tb.y);
            afr[ks][6]  = (f16)(hb.z * tb.z); afr[ks][7]  = (f16)(hb.w * tb.w);
            afr[ks][8]  = (f16)(hc.x * tc.x); afr[ks][9]  = (f16)(hc.y * tc.y);
            afr[ks][10] = (f16)(hc.z * tc.z); afr[ks][11] = (f16)(hc.w * tc.w);
            afr[ks][12] = (f16)(hd.x * td.x); afr[ks][13] = (f16)(hd.y * td.y);
            afr[ks][14] = (f16)(hd.z * td.z); afr[ks][15] = (f16)(hd.w * td.w);
        }

        // ---- dot[i,j] = row-sum of A via ones-column WMMA ----
        float dotv[8];
        {
            v8f c = {};
            #pragma unroll
            for (int ks = 0; ks < 4; ++ks)
                c = __builtin_amdgcn_wmma_f32_16x16x32_f16(
                        false, afr[ks], false, bones, (short)0, c, false, false);
            #pragma unroll
            for (int r = 0; r < 8; ++r) dotv[r] = c[r];  // valid on lanes 0 & 16
        }

        // ---- MLP: 16 N-tiles of hidden, fused bias+ReLU+W2 reduction ----
        float plog[8];
        #pragma unroll
        for (int r = 0; r < 8; ++r) plog[r] = 0.0f;

        for (int nt = 0; nt < 16; ++nt) {
            v8f c = {};
            #pragma unroll
            for (int ks = 0; ks < 4; ++ks) {
                // B fragment: W1^T[nt*16+ln][ks*32 + hs*8 + 0..7] and [ks*32 + 16 + hs*8 + 0..7]
                const uint4* bp = (const uint4*)(w1t + (nt * 16 + ln) * TWO_D
                                                 + ks * 32 + hs * 8);
                union { uint4 u[2]; v16h h; } cv;
                cv.u[0] = bp[0];
                cv.u[1] = bp[2];
                c = __builtin_amdgcn_wmma_f32_16x16x32_f16(
                        false, afr[ks], false, cv.h, (short)0, c, false, false);
            }
            const int n = nt * 16 + ln;
            const float bv = b1s[n];
            const float wv = w2s[n];
            #pragma unroll
            for (int r = 0; r < 8; ++r) {
                float h = c[r] + bv;
                h = fmaxf(h, 0.0f);
                plog[r] += h * wv;
            }
        }

        // ---- reduce partial logits across each 16-lane group ----
        #pragma unroll
        for (int r = 0; r < 8; ++r) {
            float v = plog[r];
            v += __shfl_xor(v, 1, 32);
            v += __shfl_xor(v, 2, 32);
            v += __shfl_xor(v, 4, 32);
            v += __shfl_xor(v, 8, 32);
            plog[r] = v;
        }

        // ---- masked exp, accumulate S0 / S1 (lanes 0 & 16 own 8 rows each) ----
        if (ln == 0) {
            const int mbase = pass * 128 + wid * 16 + hs * 8;
            #pragma unroll
            for (int r = 0; r < 8; ++r) {
                const int j = mbase + r;
                float e = __expf(plog[r]);
                e = (history[j] != ti) ? e : 0.0f;
                s0 += e;
                s1 += e * dotv[r];
            }
        }
    }

    // ---- workgroup reduction: 16 contributors (8 waves x 2 half-waves) ----
    if (ln == 0) {
        red[(wid * 2 + hs) * 2 + 0] = s0;
        red[(wid * 2 + hs) * 2 + 1] = s1;
    }
    __syncthreads();
    if (t == 0) {
        float S0 = 0.0f, S1 = 0.0f;
        #pragma unroll
        for (int q = 0; q < 16; ++q) { S0 += red[2 * q]; S1 += red[2 * q + 1]; }
        const float pred = S1 / sqrtf(S0);        // BETA = 0.5
        out[i] = 1.0f / (1.0f + __expf(-pred));   // sigmoid
    }
}

extern "C" void kernel_launch(void* const* d_in, const int* in_sizes, int n_in,
                              void* d_out, int out_size, void* d_ws, size_t ws_size,
                              hipStream_t stream) {
    (void)in_sizes; (void)n_in; (void)d_ws; (void)ws_size; (void)out_size;
    const int*   history        = (const int*)d_in[0];
    const int*   target         = (const int*)d_in[1];
    const int*   history_region = (const int*)d_in[2];
    const int*   target_region  = (const int*)d_in[3];
    const float* emb_history    = (const float*)d_in[4];
    const float* emb_target     = (const float*)d_in[5];
    const float* emb_region     = (const float*)d_in[6];
    const float* W1             = (const float*)d_in[7];
    const float* b1             = (const float*)d_in[8];
    const float* W2             = (const float*)d_in[9];
    float*       out            = (float*)d_out;

    nais_region_kernel<<<NB, 256, LDS_BYTES, stream>>>(
        history, target, history_region, target_region,
        emb_history, emb_target, emb_region, W1, b1, W2, out);
}
